// BinaryAttention_22041772163362
// MI455X (gfx1250) — 16-bit and 8-bit matrix operand order repaired and checked by reading, compiled, not run on hardware
//
#include <hip/hip_runtime.h>

typedef __attribute__((ext_vector_type(16))) _Float16 v16h;
typedef __attribute__((ext_vector_type(8)))  _Float16 v8h;
typedef __attribute__((ext_vector_type(8)))  float    v8f;
typedef __attribute__((ext_vector_type(8)))  int      v8i;

#define H_HEADS 12
#define NTOK    197
#define DHEAD   64
#define CDIM    768
#define BATCH   64
#define MROWS   (BATCH * NTOK)
#define BHN     (BATCH * H_HEADS)
#define NPAD    208
#define KPAD    256

__global__ __launch_bounds__(256)
void cvt_w_kernel(const float* __restrict__ w, _Float16* __restrict__ wh, int n)
{
  for (int i = blockIdx.x * 256 + threadIdx.x; i < n; i += gridDim.x * 256)
    wh[i] = (_Float16)w[i];
}

__device__ __forceinline__ void gemm32_core(const float* __restrict__ A,
                                            const _Float16* __restrict__ Wh,
                                            _Float16* aLds, v8f acc[2][4])
{
  const int tid   = threadIdx.x;
  const int lane  = tid & 31;
  const int wv    = tid >> 5;
  const int mbase = blockIdx.x * 32;
  const int n0    = blockIdx.y * 256 + wv * 64;

  for (int idx = tid; idx < 32 * CDIM; idx += 128) {
    int r = idx / CDIM;
    int c = idx - r * CDIM;
    aLds[idx] = (_Float16)A[(size_t)(mbase + r) * CDIM + c];
  }
  __syncthreads();

  const int row = lane & 15;
  const int hi  = lane >> 4;

#pragma unroll 1
  for (int kk = 0; kk < CDIM / 32; ++kk) {
    v16h a[2];
#pragma unroll
    for (int mt = 0; mt < 2; ++mt) {
      const _Float16* arow = aLds + (mt * 16 + row) * CDIM + kk * 32 + hi * 16;
      const v8h lo = *(const v8h*)(arow);
      const v8h hh = *(const v8h*)(arow + 8);
#pragma unroll
      for (int p = 0; p < 8; ++p) { a[mt][p] = lo[p]; a[mt][p + 8] = hh[p]; }
    }
#pragma unroll
    for (int t = 0; t < 4; ++t) {
      const int n = n0 + t * 16 + (lane & 15);
      const v16h bf = *(const v16h*)(Wh + (size_t)n * CDIM + kk * 32 + hi * 16);
#pragma unroll
      for (int mt = 0; mt < 2; ++mt) {
        acc[mt][t] = __builtin_amdgcn_wmma_f32_16x16x32_f16(
            false, a[mt], false, bf, (short)0, acc[mt][t], false, false);
      }
    }
  }
}

__global__ __launch_bounds__(128)
void qkv_gemm_kernel(const float* __restrict__ x,
                     const _Float16* __restrict__ qkv_wh,
                     float* __restrict__ qkvf)
{
  __shared__ _Float16 aLds[32 * CDIM];
  v8f acc[2][4] = {};
  gemm32_core(x, qkv_wh, aLds, acc);

  const int lane  = threadIdx.x & 31;
  const int wv    = threadIdx.x >> 5;
  const int hi    = lane >> 4;
  const int mbase = blockIdx.x * 32;
  const int n0    = blockIdx.y * 256 + wv * 64;

#pragma unroll
  for (int t = 0; t < 4; ++t) {
    const int c     = n0 + t * 16 + (lane & 15);
    const int which = c / CDIM;
    const int rem   = c - which * CDIM;
    const int h     = rem >> 6;
    const int dd    = rem & 63;
#pragma unroll
    for (int mt = 0; mt < 2; ++mt) {
#pragma unroll
      for (int r = 0; r < 8; ++r) {
        const int m    = mbase + mt * 16 + r + hi * 8;
        const int b    = m / NTOK;
        const int ntok = m - b * NTOK;
        qkvf[(size_t)(which * BHN + b * H_HEADS + h) * (NTOK * DHEAD) +
             ntok * DHEAD + dd] = acc[mt][t][r];
      }
    }
  }
}

__global__ __launch_bounds__(256)
void quant_kernel(const float* __restrict__ qkvf, float* __restrict__ sc,
                  signed char* __restrict__ qb, signed char* __restrict__ kb,
                  signed char* __restrict__ vb)
{
  __shared__ float red[256];
  const int bh  = blockIdx.x;
  const int tid = threadIdx.x;
  const size_t off = (size_t)bh * (NTOK * DHEAD);
  const float* q = qkvf + off;
  const float* k = qkvf + (size_t)BHN * (NTOK * DHEAD) + off;
  const float* v = qkvf + (size_t)2 * BHN * (NTOK * DHEAD) + off;

  float sq = 0.f, sk = 0.f;
  for (int i = tid; i < NTOK * DHEAD; i += 256) {
    const float qv = q[i], kv = k[i], vv = v[i];
    sq += fabsf(qv);
    sk += fabsf(kv);
    qb[off + i] = (qv >= 0.f) ? (signed char)1 : (signed char)-1;
    kb[off + i] = (kv >= 0.f) ? (signed char)1 : (signed char)-1;
    float vc = fminf(fmaxf(vv, -2.f), 2.f);
    float rq = rintf(vc * 63.5f);
    rq = fminf(fmaxf(rq, -127.f), 127.f);
    vb[off + i] = (signed char)rq;
  }

  red[tid] = sq; __syncthreads();
  for (int s = 128; s > 0; s >>= 1) { if (tid < s) red[tid] += red[tid + s]; __syncthreads(); }
  if (tid == 0) sc[bh] = red[0] * (1.0f / (NTOK * DHEAD));
  __syncthreads();
  red[tid] = sk; __syncthreads();
  for (int s = 128; s > 0; s >>= 1) { if (tid < s) red[tid] += red[tid + s]; __syncthreads(); }
  if (tid == 0) sc[BHN + bh] = red[0] * (1.0f / (NTOK * DHEAD));
}

__global__ __launch_bounds__(128)
void attn_kernel(const signed char* __restrict__ qb,
                 const signed char* __restrict__ kb,
                 const signed char* __restrict__ vb,
                 const float* __restrict__ sc,
                 const float* __restrict__ rel_table,
                 const int* __restrict__ rel_index,
                 float* __restrict__ ctx)
{
  __shared__ alignas(16) signed char   qsm[NPAD * DHEAD];
  __shared__ alignas(16) signed char   ksm[NPAD * DHEAD];
  __shared__ alignas(16) signed char   vsmT[DHEAD * KPAD];
  __shared__ alignas(16) unsigned char psm[4 * 16 * KPAD];

  const int bh   = blockIdx.x;
  const int b    = bh / H_HEADS;
  const int h    = bh - b * H_HEADS;
  const int tid  = threadIdx.x;
  const int lane = tid & 31;
  const int wv   = tid >> 5;
  const float coeff = sc[bh] * sc[BHN + bh] * 0.125f;

  const size_t base8 = (size_t)bh * (NTOK * DHEAD);
  for (int i = tid; i < NPAD * DHEAD; i += 128) {
    signed char qv = 0, kv = 0;
    if (i < NTOK * DHEAD) { qv = qb[base8 + i]; kv = kb[base8 + i]; }
    qsm[i] = qv; ksm[i] = kv;
  }
  for (int i = tid; i < DHEAD * KPAD; i += 128) {
    const int dd  = i >> 8;
    const int tok = i & 255;
    vsmT[i] = (tok < NTOK) ? vb[base8 + (size_t)tok * DHEAD + dd] : (signed char)0;
  }
  unsigned char* psm_w = psm + wv * (16 * KPAD);
  {
    int* pz = (int*)psm_w;
    for (int i = lane; i < (16 * KPAD) / 4; i += 32) pz[i] = 0;
  }
  __syncthreads();

  const int lhalf = lane & 15;
  const int hi    = lane >> 4;

  for (int it = wv; it < 13; it += 4) {
    const int r0 = it * 16;

    v8i aq;
    {
      const signed char* qrow = qsm + (r0 + lhalf) * DHEAD + hi * 16;
#pragma unroll
      for (int p = 0; p < 8; ++p) {
        const int off = ((p >> 2) << 5) + ((p & 3) << 2);
        aq[p] = *(const int*)(qrow + off);
      }
    }

    float mrow[8], lrow[8];
#pragma unroll
    for (int r = 0; r < 8; ++r) { mrow[r] = -3.0e38f; lrow[r] = 0.f; }

    for (int jt = 0; jt < 13; ++jt) {
      v8i bk;
      const signed char* krow = ksm + (jt * 16 + lhalf) * DHEAD + hi * 16;
#pragma unroll
      for (int p = 0; p < 8; ++p) {
        const int off = ((p >> 2) << 5) + ((p & 3) << 2);
        bk[p] = *(const int*)(krow + off);
      }
      v8i zero = {};
      v8i s = __builtin_amdgcn_wmma_i32_16x16x64_iu8(true, aq, true, bk, zero,
                                                     false, false);
      const int nc = jt * 16 + lhalf;
#pragma unroll
      for (int r = 0; r < 8; ++r) {
        if (nc < NTOK) {
          const int nr  = r0 + r + hi * 8;
          const int idx = (nr < NTOK) ? rel_index[nr * NTOK + nc] : 0;
          const float val = coeff * (float)s[r] + rel_table[idx * H_HEADS + h];
          const float mo = mrow[r];
          const float mn = fmaxf(mo, val);
          lrow[r] = lrow[r] * __expf(mo - mn) + __expf(val - mn);
          mrow[r] = mn;
        }
      }
    }
#pragma unroll
    for (int r = 0; r < 8; ++r) {
      float m = mrow[r], l = lrow[r];
#pragma unroll
      for (int off = 1; off < 16; off <<= 1) {
        const float m2 = __shfl_xor(m, off);
        const float l2 = __shfl_xor(l, off);
        const float mn = fmaxf(m, m2);
        l = l * __expf(m - mn) + l2 * __expf(m2 - mn);
        m = mn;
      }
      mrow[r] = m;
      lrow[r] = 255.0f / l;
    }

    for (int jt = 0; jt < 13; ++jt) {
      v8i bk;
      const signed char* krow = ksm + (jt * 16 + lhalf) * DHEAD + hi * 16;
#pragma unroll
      for (int p = 0; p < 8; ++p) {
        const int off = ((p >> 2) << 5) + ((p & 3) << 2);
        bk[p] = *(const int*)(krow + off);
      }
      v8i zero = {};
      v8i s = __builtin_amdgcn_wmma_i32_16x16x64_iu8(true, aq, true, bk, zero,
                                                     false, false);
      const int nc = jt * 16 + lhalf;
#pragma unroll
      for (int r = 0; r < 8; ++r) {
        unsigned char pq = 0;
        if (nc < NTOK) {
          const int nr  = r0 + r + hi * 8;
          const int idx = (nr < NTOK) ? rel_index[nr * NTOK + nc] : 0;
          const float val = coeff * (float)s[r] + rel_table[idx * H_HEADS + h];
          float p = __expf(val - mrow[r]) * lrow[r];
          p = fminf(fmaxf(rintf(p), 0.f), 255.f);
          pq = (unsigned char)p;
        }
        psm_w[(r + hi * 8) * KPAD + nc] = pq;
      }
    }

    const float OUT_SCALE = 2.0f / (255.0f * 127.0f);
#pragma unroll 1
    for (int jt2 = 0; jt2 < 4; ++jt2) {
      v8i acc = {};
#pragma unroll
      for (int kc = 0; kc < 4; ++kc) {
        v8i ap, bv2;
        const unsigned char* prow = psm_w + lhalf * KPAD + kc * 64 + hi * 16;
#pragma unroll
        for (int p = 0; p < 8; ++p) {
          const int off = ((p >> 2) << 5) + ((p & 3) << 2);
          ap[p] = *(const int*)(prow + off);
        }
        const signed char* vrow = vsmT + (jt2 * 16 + lhalf) * KPAD + kc * 64 + hi * 16;
#pragma unroll
        for (int p = 0; p < 8; ++p) {
          const int off = ((p >> 2) << 5) + ((p & 3) << 2);
          bv2[p] = *(const int*)(vrow + off);
        }
        acc = __builtin_amdgcn_wmma_i32_16x16x64_iu8(false, ap, true, bv2, acc,
                                                     false, false);
      }
#pragma unroll
      for (int r = 0; r < 8; ++r) {
        const int tok = r0 + r + hi * 8;
        if (tok < NTOK) {
          ctx[((size_t)b * NTOK + tok) * CDIM + h * DHEAD + jt2 * 16 + lhalf] =
              (float)acc[r] * OUT_SCALE;
        }
      }
    }
  }
}

__global__ __launch_bounds__(128)
void proj_gemm_kernel(const float* __restrict__ ctx,
                      const _Float16* __restrict__ proj_wh,
                      const float* __restrict__ proj_b,
                      float* __restrict__ out)
{
  __shared__ _Float16 aLds[32 * CDIM];
  v8f acc[2][4] = {};
  gemm32_core(ctx, proj_wh, aLds, acc);

  const int lane  = threadIdx.x & 31;
  const int wv    = threadIdx.x >> 5;
  const int hi    = lane >> 4;
  const int mbase = blockIdx.x * 32;
  const int n0    = blockIdx.y * 256 + wv * 64;

#pragma unroll
  for (int t = 0; t < 4; ++t) {
    const int c = n0 + t * 16 + (lane & 15);
    const float bias = proj_b[c];
#pragma unroll
    for (int mt = 0; mt < 2; ++mt) {
#pragma unroll
      for (int r = 0; r < 8; ++r) {
        const int m = mbase + mt * 16 + r + hi * 8;
        out[(size_t)m * CDIM + c] = acc[mt][t][r] + bias;
      }
    }
  }
}

extern "C" void kernel_launch(void* const* d_in, const int* in_sizes, int n_in,
                              void* d_out, int out_size, void* d_ws, size_t ws_size,
                              hipStream_t stream)
{
  const float* x         = (const float*)d_in[0];
  const float* qkv_w     = (const float*)d_in[1];
  const float* proj_w    = (const float*)d_in[2];
  const float* proj_b    = (const float*)d_in[3];
  const float* rel_table = (const float*)d_in[4];
  const int*   rel_index = (const int*)d_in[5];

  float* qkvf = (float*)d_ws;
  float* ctx  = qkvf + (size_t)3 * BHN * NTOK * DHEAD;
  float* sc   = ctx + (size_t)MROWS * CDIM;
  signed char* qb = (signed char*)(sc + 2 * BHN);
  signed char* kb = qb + (size_t)BHN * NTOK * DHEAD;
  signed char* vb = kb + (size_t)BHN * NTOK * DHEAD;
  _Float16* qkv_wh  = (_Float16*)(vb + (size_t)BHN * NTOK * DHEAD);
  _Float16* proj_wh = qkv_wh + (size_t)3 * CDIM * CDIM;

  hipLaunchKernelGGL(cvt_w_kernel, dim3(864), dim3(256), 0, stream,
                     qkv_w, qkv_wh, 3 * CDIM * CDIM);
  hipLaunchKernelGGL(cvt_w_kernel, dim3(288), dim3(256), 0, stream,
                     proj_w, proj_wh, CDIM * CDIM);

  dim3 g1(MROWS / 32, (3 * CDIM) / 256);
  hipLaunchKernelGGL(qkv_gemm_kernel, g1, dim3(128), 0, stream, x, qkv_wh, qkvf);

  hipLaunchKernelGGL(quant_kernel, dim3(BHN), dim3(256), 0, stream,
                     qkvf, sc, qb, kb, vb);

  hipLaunchKernelGGL(attn_kernel, dim3(BHN), dim3(128), 0, stream,
                     qb, kb, vb, sc, rel_table, rel_index, ctx);

  dim3 g4(MROWS / 32, CDIM / 256);
  hipLaunchKernelGGL(proj_gemm_kernel, g4, dim3(128), 0, stream,
                     ctx, proj_wh, proj_b, (float*)d_out);
}
